// RandomLinear_47304769798217
// MI455X (gfx1250) — matrix operand order repaired and checked by reading, not compiled, not run on hardware
//
#include <hip/hip_runtime.h>

typedef __attribute__((ext_vector_type(16))) _Float16 v16h;
typedef __attribute__((ext_vector_type(8)))  _Float16 v8h;
typedef __attribute__((ext_vector_type(8)))  float    v8f;
typedef __attribute__((ext_vector_type(4)))  float    v4f;

#define KDIM 2048
#define NDIM 2048
#define WIN  64            // mask block size
#define LDSS 72            // padded LDS row stride (f16 elems)

__global__ __launch_bounds__(256) void rlin_wmma_kernel(
    const float* __restrict__ x,     // [M, K]
    const float* __restrict__ w,     // [N, K]
    const float* __restrict__ mask,  // [N, K], 0/1 in 64x64 blocks
    const float* __restrict__ bias,  // [N]
    float* __restrict__ out)         // [M, N]
{
    __shared__ __attribute__((aligned(16))) _Float16 Bs[WIN][LDSS];

    const int tid  = threadIdx.x;
    const int wave = tid >> 5;        // 0..7  (wave32)
    const int lane = tid & 31;
    const int half = (lane >> 4) & 1; // selects K half per ISA fragment layout
    const int l16  = lane & 15;

    const int m_wave = blockIdx.x * 256 + wave * 32;  // 32 M-rows per wave
    const int n_base = blockIdx.y * WIN;              // 64 N-cols per block

    v8f acc[2][4];
#pragma unroll
    for (int mt = 0; mt < 2; ++mt)
#pragma unroll
        for (int nt = 0; nt < 4; ++nt) {
            v8f z = {};
            acc[mt][nt] = z;
        }

    // per-lane A row pointers (A fragment: lane l16 carries row m_tile+l16)
    const float* arow0 = x + (size_t)(m_wave +      l16) * KDIM;
    const float* arow1 = x + (size_t)(m_wave + 16 + l16) * KDIM;

    // cooperative W staging: 64 rows x 64 f32 -> f16 LDS; 4 threads per row
    const int wrow = tid >> 2;          // 0..63
    const int wseg = (tid & 3) * 16;    // 0,16,32,48 (f16 elems)

    for (int k0 = 0; k0 < KDIM; k0 += WIN) {
        // Block-uniform sparsity check on the 64x64 mask block corner.
        if (mask[(size_t)n_base * KDIM + (size_t)k0] == 0.0f)
            continue;  // uniform across block: barriers below stay legal

        // ---- stage W[n_base..+64, k0..+64] into LDS as f16 ----
        {
            const float* wp = w + (size_t)(n_base + wrow) * KDIM + k0 + wseg;
            v4f f0 = *(const v4f*)(wp + 0);
            v4f f1 = *(const v4f*)(wp + 4);
            v4f f2 = *(const v4f*)(wp + 8);
            v4f f3 = *(const v4f*)(wp + 12);
            v8h h0, h1;
#pragma unroll
            for (int j = 0; j < 4; ++j) {
                h0[j]     = (_Float16)f0[j];
                h0[4 + j] = (_Float16)f1[j];
                h1[j]     = (_Float16)f2[j];
                h1[4 + j] = (_Float16)f3[j];
            }
            *(v8h*)&Bs[wrow][wseg + 0] = h0;
            *(v8h*)&Bs[wrow][wseg + 8] = h1;
        }
        __syncthreads();

        // prefetch next chunk of X rows (global_prefetch_b8 path)
        if (k0 + WIN < KDIM) {
            __builtin_prefetch(arow0 + k0 + WIN, 0, 0);
            __builtin_prefetch(arow1 + k0 + WIN, 0, 0);
        }

        // ---- two K=32 WMMA steps per 64-chunk ----
#pragma unroll
        for (int kk = 0; kk < WIN; kk += 32) {
            // A fragment offsets per ISA: lane<16 -> K {0..7, 16..23}; +8 for upper half
            const int oa0 = kk + half * 8;
            const int oa1 = kk + 16 + half * 8;

            v16h af[2];
            {
                const float* aps[2] = { arow0 + k0, arow1 + k0 };
#pragma unroll
                for (int mt = 0; mt < 2; ++mt) {
                    v4f a0 = *(const v4f*)(aps[mt] + oa0);
                    v4f a1 = *(const v4f*)(aps[mt] + oa0 + 4);
                    v4f a2 = *(const v4f*)(aps[mt] + oa1);
                    v4f a3 = *(const v4f*)(aps[mt] + oa1 + 4);
                    v16h t;
#pragma unroll
                    for (int j = 0; j < 4; ++j) {
                        t[j]      = (_Float16)a0[j];
                        t[4 + j]  = (_Float16)a1[j];
                        t[8 + j]  = (_Float16)a2[j];
                        t[12 + j] = (_Float16)a3[j];
                    }
                    af[mt] = t;
                }
            }

            // Preload ALL four B fragments so the 8 ds_load_b128s issue as one
            // clause and the 8 WMMAs below run without per-pair dscnt stalls.
            // B fragment matches A: lane<16 -> K {0..7, 16..23}; +8 for upper half
            const int ob = kk + half * 8;
            v16h bf[4];
#pragma unroll
            for (int nt = 0; nt < 4; ++nt) {
                v8h b0 = *(const v8h*)&Bs[nt * 16 + l16][ob + 0];
                v8h b1 = *(const v8h*)&Bs[nt * 16 + l16][ob + 16];
                v16h t;
#pragma unroll
                for (int j = 0; j < 8; ++j) { t[j] = b0[j]; t[8 + j] = b1[j]; }
                bf[nt] = t;
            }

#pragma unroll
            for (int nt = 0; nt < 4; ++nt)
#pragma unroll
                for (int mt = 0; mt < 2; ++mt) {
                    acc[mt][nt] = __builtin_amdgcn_wmma_f32_16x16x32_f16(
                        false, af[mt], false, bf[nt],
                        (short)0, acc[mt][nt], false, false);
                }
        }
        __syncthreads();  // WAR guard before next chunk restages Bs
    }

    // ---- epilogue: bias + store (C/D layout: lane<16 M0-7, lane>=16 M8-15) ----
    const int rsel = half * 8;
#pragma unroll
    for (int nt = 0; nt < 4; ++nt) {
        const int col = n_base + nt * 16 + l16;
        const float bv = bias[col];
#pragma unroll
        for (int mt = 0; mt < 2; ++mt) {
            float* cp = out + (size_t)(m_wave + mt * 16 + rsel) * NDIM + col;
#pragma unroll
            for (int i = 0; i < 8; ++i)
                cp[(size_t)i * NDIM] = acc[mt][nt][i] + bv;
        }
    }
}

extern "C" void kernel_launch(void* const* d_in, const int* in_sizes, int n_in,
                              void* d_out, int out_size, void* d_ws, size_t ws_size,
                              hipStream_t stream) {
    (void)n_in; (void)out_size; (void)d_ws; (void)ws_size;
    const float* x    = (const float*)d_in[0];  // [4,8192,2048] -> [M,K]
    const float* wgt  = (const float*)d_in[1];  // [2048,2048]   -> [N,K]
    const float* mask = (const float*)d_in[2];  // [2048,2048]
    const float* bias = (const float*)d_in[3];  // [2048]
    float* out = (float*)d_out;

    const int M = in_sizes[0] / KDIM;           // 32768
    dim3 grid((unsigned)(M / 256), (unsigned)(NDIM / WIN));  // 128 x 32
    rlin_wmma_kernel<<<grid, 256, 0, stream>>>(x, wgt, mask, bias, out);
}
